// SALAD_73899207295305
// MI455X (gfx1250) — matrix operand order repaired and checked by reading, not compiled, not run on hardware
//
#include <hip/hip_runtime.h>
#include <hip/hip_bf16.h>

// ---------------------------------------------------------------------------
// MI455X (gfx1250) SALAD pipeline.
// All GEMM-shaped work (2x C->512 convs ~103 GFLOP, 2nd-stage convs, burst
// similarity fn^T fn, VLAD aggregation f P^T) runs on v_wmma_f32_16x16x32_f16
// (wave32 WMMA). f32 inputs are converted to f16 while staging tiles in LDS
// via vectorized b128 loads; f16 B tiles use GLOBAL_LOAD_ASYNC_TO_LDS_B128
// (ASYNCcnt) when the toolchain exposes the builtin.
// ---------------------------------------------------------------------------

typedef __attribute__((ext_vector_type(16))) _Float16 v16h;
typedef __attribute__((ext_vector_type(8)))  _Float16 v8h;
typedef __attribute__((ext_vector_type(8)))  float    v8f;

#if defined(__gfx1250__) && __has_builtin(__builtin_amdgcn_global_load_async_to_lds_b128)
#define SALAD_HAS_ASYNC_LDS 1
// Builtin signature (from hipcc diagnostic): param0 is a non-const pointer to
// a gcc-style int 4-vector in the global (AS1) address space; param1 is the
// LDS (AS3) destination; params 2/3 are immediate offset / cache policy.
typedef int salad_v4i __attribute__((vector_size(16)));
typedef __attribute__((address_space(1))) salad_v4i* salad_gbl_v4i_p;
typedef __attribute__((address_space(3))) salad_v4i* salad_lds_v4i_p;
#else
#define SALAD_HAS_ASYNC_LDS 0
#endif

// Build a 16x32 f16 A-fragment per the CDNA5 ISA layout:
//  lane<16 : row M=lane,    elems 0..7 = K[0..7],  elems 8..15 = K[16..23]
//  lane>=16: row M=lane-16, elems 0..7 = K[8..15], elems 8..15 = K[24..31]
__device__ inline v16h make_afrag(const _Float16* rowp, int khalf) {
  v8h lo = *(const v8h*)(rowp + khalf * 8);
  v8h hi = *(const v8h*)(rowp + 16 + khalf * 8);
  v16h a;
#pragma unroll
  for (int i = 0; i < 8; ++i) { a[i] = lo[i]; a[8 + i] = hi[i]; }
  return a;
}

// ---------------------------------------------------------------------------
// Generic batched WMMA GEMM:  out[b] = act(A[b] (MxK) * B[b] (KxN) + bias)
// A row-major lda=K, B row-major ldb=N. 256 threads = 8 waves; wave tile
// 16x64 (4 accumulators); workgroup tile BM x BN, (BM/16)*(BN/64)==8. KT=32.
// ---------------------------------------------------------------------------
template <int BM, int BN, bool RELU, typename TA, typename TB>
__global__ __launch_bounds__(256) void gemm_wmma(
    const TA* __restrict__ A, unsigned long long strideAb,
    const TB* __restrict__ B, unsigned long long strideBb,
    const float* __restrict__ bias,
    float* __restrict__ outF, _Float16* __restrict__ outH,
    int M, int N, int K) {
  constexpr int WR = BM / 16;
  constexpr int WC = BN / 64;
  static_assert(WR * WC == 8, "need 8 waves");
  __shared__ __align__(32) _Float16 sA[BM * 32];
  __shared__ __align__(32) _Float16 sB[32 * BN];

  const int tid  = threadIdx.x;
  const int lane = tid & 31;
  const int wave = tid >> 5;
  const int wrow = wave % WR;
  const int wcol = wave / WR;
  const int half = lane >> 4;
  const int lrow = lane & 15;

  const int n0 = blockIdx.x * BN;
  const int m0 = blockIdx.y * BM;
  const int b  = blockIdx.z;
  A += (size_t)b * strideAb;
  B += (size_t)b * strideBb;

  v8f acc[4] = {};

  const int nK = K / 32;
  for (int kt = 0; kt < nK; ++kt) {
    const int k0 = kt * 32;
    if (kt + 1 < nK) {  // hint next K tile -> global_prefetch_b8
      __builtin_prefetch(&A[(size_t)m0 * K + k0 + 32], 0, 1);
      __builtin_prefetch(&B[(size_t)(k0 + 32) * N + n0], 0, 1);
    }
    __syncthreads();

    // ---- stage A tile (BM x 32), convert to f16. float4-vectorized. ----
    if constexpr (__hip_internal::is_same<TA, float>::value) {
#pragma unroll
      for (int vv = tid; vv < BM * 8; vv += 256) {  // BM*32/4 float4 chunks
        const int r = vv >> 3, c4 = (vv & 7) * 4;
        const float4 q = *(const float4*)&A[(size_t)(m0 + r) * K + k0 + c4];
        _Float16* d = &sA[r * 32 + c4];
        d[0] = (_Float16)q.x; d[1] = (_Float16)q.y;
        d[2] = (_Float16)q.z; d[3] = (_Float16)q.w;
      }
    } else {
#pragma unroll
      for (int vv = tid; vv < BM * 4; vv += 256) {  // BM*32/8 v8h chunks
        const int r = vv >> 2, c8 = (vv & 3) * 8;
        *(v8h*)&sA[r * 32 + c8] =
            *(const v8h*)&A[(size_t)(m0 + r) * K + k0 + c8];
      }
    }

    // ---- stage B tile (32 x BN) ----
    if constexpr (__hip_internal::is_same<TB, float>::value) {
#pragma unroll
      for (int vv = tid; vv < 8 * BN; vv += 256) {  // 32*BN/4 float4 chunks
        const int r = vv / (BN / 4), c4 = (vv % (BN / 4)) * 4;
        const float4 q = *(const float4*)&B[(size_t)(k0 + r) * N + n0 + c4];
        _Float16* d = &sB[r * BN + c4];
        d[0] = (_Float16)q.x; d[1] = (_Float16)q.y;
        d[2] = (_Float16)q.z; d[3] = (_Float16)q.w;
      }
    } else {
      // f16 B tile is a raw byte copy: use async global->LDS b128 if present.
#if SALAD_HAS_ASYNC_LDS
#pragma unroll
      for (int vv = tid; vv < 4 * BN; vv += 256) {  // 32*BN/8 16-byte chunks
        const int r = vv / (BN / 8), c8 = (vv % (BN / 8)) * 8;
        const TB* gsrc = &B[(size_t)(k0 + r) * N + n0 + c8];
        _Float16* ldst = &sB[r * BN + c8];
        __builtin_amdgcn_global_load_async_to_lds_b128(
            (salad_gbl_v4i_p)(void*)gsrc, (salad_lds_v4i_p)(void*)ldst, 0, 0);
      }
      asm volatile("s_wait_asynccnt 0x0" ::: "memory");
#else
#pragma unroll
      for (int vv = tid; vv < 4 * BN; vv += 256) {
        const int r = vv / (BN / 8), c8 = (vv % (BN / 8)) * 8;
        *(v8h*)&sB[r * BN + c8] =
            *(const v8h*)&B[(size_t)(k0 + r) * N + n0 + c8];
      }
#endif
    }
    __syncthreads();

    const v16h a = make_afrag(&sA[(wrow * 16 + lrow) * 32], half);
#pragma unroll
    for (int s = 0; s < 4; ++s) {
      // B frag: lane holds N-col `lrow`, 16 K values in table order.
      v16h bb;
#pragma unroll
      for (int i = 0; i < 16; ++i)
        bb[i] = sB[((i < 8) ? (half * 8 + i) : (16 + half * 8 + i - 8)) * BN + wcol * 64 + s * 16 + lrow];
      acc[s] = __builtin_amdgcn_wmma_f32_16x16x32_f16(
          false, a, false, bb, (short)0, acc[s], false, false);
    }
  }

  // C/D layout: VGPR r -> M = r + 8*half, N = lrow (within 16-wide tile).
#pragma unroll
  for (int s = 0; s < 4; ++s) {
    const int gn = n0 + wcol * 64 + s * 16 + lrow;
#pragma unroll
    for (int r = 0; r < 8; ++r) {
      const int gm = m0 + wrow * 16 + half * 8 + r;
      float vl = acc[s][r];
      if (bias) vl += bias[gm];
      if (RELU) vl = fmaxf(vl, 0.f);
      const size_t oi = (size_t)b * M * N + (size_t)gm * N + gn;
      if (outF) outF[oi] = vl;
      if (outH) outH[oi] = (_Float16)vl;
    }
  }
}

// ---------------------------------------------------------------------------
// Token MLP: tk[b] = relu(t[b] @ Wt1^T + bt1) @ Wt2^T + bt2   (tiny, VALU)
// ---------------------------------------------------------------------------
__global__ __launch_bounds__(256) void token_mlp(
    const float* __restrict__ t, const float* __restrict__ Wt1,
    const float* __restrict__ bt1, const float* __restrict__ Wt2,
    const float* __restrict__ bt2, float* __restrict__ tk) {
  const int b = blockIdx.x, tid = threadIdx.x;
  __shared__ float tl[1536];
  __shared__ float hid[512];
  for (int e = tid; e < 1536; e += 256) tl[e] = t[b * 1536 + e];
  __syncthreads();
  for (int o = tid; o < 512; o += 256) {
    const float* row = Wt1 + (size_t)o * 1536;
    float d = bt1[o];
    for (int c = 0; c < 1536; ++c) d += row[c] * tl[c];
    hid[o] = fmaxf(d, 0.f);
  }
  __syncthreads();
  {
    const int g = tid;
    const float* row = Wt2 + (size_t)g * 512;
    float d = bt2[g];
    for (int c = 0; c < 512; ++c) d += row[c] * hid[c];
    tk[b * 256 + g] = d;
  }
}

// ---------------------------------------------------------------------------
// Sinkhorn (log space, 3 iters). One block per batch. Z rows 0..63 = p,
// row 64 = dust_bin. u[65] and v[1024] kept in LDS, written out at the end.
// ---------------------------------------------------------------------------
__global__ __launch_bounds__(256) void sinkhorn_kernel(
    const float* __restrict__ p, const float* __restrict__ dustp,
    float* __restrict__ uOut, float* __restrict__ vOut) {
  const int b = blockIdx.x, tid = threadIdx.x;
  const int lane = tid & 31, wave = tid >> 5;
  __shared__ float su[80];
  __shared__ float sv[1024];
  const float dust = *dustp;
  const float norm = -logf(1088.f);            // -log(m+n)
  const float lmu_last = logf(1024.f) + norm;  // log(n)+norm
  const float* Z = p + (size_t)b * 65536;

  for (int j = tid; j < 1024; j += 256) sv[j] = 0.f;
  __syncthreads();

  for (int it = 0; it < 3; ++it) {
    for (int i = wave; i < 65; i += 8) {
      const float* zr = (i < 64) ? (Z + (size_t)i * 1024) : nullptr;
      float mx = -3.4e38f;
      for (int j = lane; j < 1024; j += 32) {
        const float z = zr ? zr[j] : dust;
        mx = fmaxf(mx, z + sv[j]);
      }
      for (int m = 16; m; m >>= 1) mx = fmaxf(mx, __shfl_xor(mx, m, 32));
      float s = 0.f;
      for (int j = lane; j < 1024; j += 32) {
        const float z = zr ? zr[j] : dust;
        s += expf(z + sv[j] - mx);
      }
      for (int m = 16; m; m >>= 1) s += __shfl_xor(s, m, 32);
      if (lane == 0) {
        const float lmu = (i < 64) ? norm : lmu_last;
        su[i] = lmu - (mx + logf(s));
      }
    }
    __syncthreads();
    for (int j = tid; j < 1024; j += 256) {
      float mx = -3.4e38f;
      for (int i = 0; i < 65; ++i) {
        const float z = (i < 64) ? Z[(size_t)i * 1024 + j] : dust;
        mx = fmaxf(mx, z + su[i]);
      }
      float s = 0.f;
      for (int i = 0; i < 65; ++i) {
        const float z = (i < 64) ? Z[(size_t)i * 1024 + j] : dust;
        s += expf(z + su[i] - mx);
      }
      sv[j] = norm - (mx + logf(s));
    }
    __syncthreads();
  }
  for (int i = tid; i < 65; i += 256) uOut[b * 65 + i] = su[i];
  for (int j = tid; j < 1024; j += 256) vOut[b * 1024 + j] = sv[j];
}

// ---------------------------------------------------------------------------
// Column-wise L2 normalize f over l(=128); emit fn (128x1024) and fnT
// (1024x128) in f16 for the burst WMMA kernel.
// ---------------------------------------------------------------------------
__global__ __launch_bounds__(256) void normalize_f(
    const float* __restrict__ f, _Float16* __restrict__ fn,
    _Float16* __restrict__ fnT) {
  const int b = blockIdx.y;
  const int j = blockIdx.x * 256 + threadIdx.x;
  const float* base = f + (size_t)b * 131072 + j;
  float ss = 0.f;
  for (int l = 0; l < 128; ++l) { const float x = base[(size_t)l * 1024]; ss += x * x; }
  const float rn = 1.f / fmaxf(sqrtf(ss), 1e-12f);
  for (int l = 0; l < 128; ++l) {
    const float x = base[(size_t)l * 1024] * rn;
    fn[(size_t)b * 131072 + (size_t)l * 1024 + j] = (_Float16)x;
    fnT[(size_t)b * 131072 + (size_t)j * 128 + l] = (_Float16)x;
  }
}

// ---------------------------------------------------------------------------
// Burst reweighting: bw[j] = sum_k sigmoid(a * (fn^T fn)[j,k] + b).
// Fused WMMA over the 1024x1024 similarity (never materialized).
// ---------------------------------------------------------------------------
__global__ __launch_bounds__(256) void burst_kernel(
    const _Float16* __restrict__ fn, const _Float16* __restrict__ fnT,
    const float* __restrict__ pa, const float* __restrict__ pb,
    float* __restrict__ bw) {
  const int b = blockIdx.y;
  const int tid = threadIdx.x, lane = tid & 31, wave = tid >> 5;
  const int j0 = (blockIdx.x * 8 + wave) * 16;
  const int half = lane >> 4;
  const float Ap = *pa, Bp = *pb;

  const _Float16* At = fnT + (size_t)b * 131072;
  const _Float16* Bn = fn + (size_t)b * 131072;

  v16h a[4];
  const int arow = j0 + (lane & 15);
#pragma unroll
  for (int ks = 0; ks < 4; ++ks)
    a[ks] = make_afrag(At + (size_t)arow * 128 + ks * 32, half);

  float rs[8] = {0.f, 0.f, 0.f, 0.f, 0.f, 0.f, 0.f, 0.f};
  for (int kt = 0; kt < 64; ++kt) {
    const int n0 = kt * 16;
    v8f acc = {};
#pragma unroll
    for (int ks = 0; ks < 4; ++ks) {
      v16h bb;
#pragma unroll
      for (int i = 0; i < 16; ++i)
        bb[i] = Bn[(size_t)(ks * 32 + ((i < 8) ? (half * 8 + i) : (16 + half * 8 + i - 8))) * 1024 + n0 + (lane & 15)];
      acc = __builtin_amdgcn_wmma_f32_16x16x32_f16(
          false, a[ks], false, bb, (short)0, acc, false, false);
    }
#pragma unroll
    for (int r = 0; r < 8; ++r) {
      const float sv = Ap * acc[r] + Bp;
      rs[r] += 1.f / (1.f + expf(-sv));
    }
  }
#pragma unroll
  for (int r = 0; r < 8; ++r) {
    float s = rs[r];
    s += __shfl_xor(s, 1, 32);
    s += __shfl_xor(s, 2, 32);
    s += __shfl_xor(s, 4, 32);
    s += __shfl_xor(s, 8, 32);
    rs[r] = s;
  }
  if ((lane & 15) == 0) {
#pragma unroll
    for (int r = 0; r < 8; ++r)
      bw[(size_t)b * 1024 + j0 + half * 8 + r] = rs[r];
  }
}

// ---------------------------------------------------------------------------
// P = exp(Z + u + v - norm) / bw^burst_p for rows 0..63, stored TRANSPOSED
// (Pt[j][m], f16) so the agg GEMM reads it as a plain row-major B matrix.
// ---------------------------------------------------------------------------
__global__ __launch_bounds__(256) void finalize_P(
    const float* __restrict__ p, const float* __restrict__ u,
    const float* __restrict__ v, const float* __restrict__ bw,
    const float* __restrict__ pp, _Float16* __restrict__ Pt) {
  const int b = blockIdx.y;
  const int j = blockIdx.x * 256 + threadIdx.x;
  const float norm = -logf(1088.f);
  const float vj = v[b * 1024 + j];
  const float ws = powf(bw[b * 1024 + j], -(*pp));
  const float* Z = p + (size_t)b * 65536 + j;
  const float* ub = u + b * 65;
  _Float16* out = Pt + (size_t)b * 65536 + (size_t)j * 64;
  for (int i = 0; i < 64; ++i)
    out[i] = (_Float16)(expf(Z[(size_t)i * 1024] + ub[i] + vj - norm) * ws);
}

// ---------------------------------------------------------------------------
// Final: colwise-normalize agg over l, normalize tk, concat, global L2 norm.
// ---------------------------------------------------------------------------
__device__ inline float block_reduce_sum(float v, float* red) {
  __syncthreads();
  for (int m = 16; m; m >>= 1) v += __shfl_xor(v, m, 32);
  const int w = threadIdx.x >> 5;
  if ((threadIdx.x & 31) == 0) red[w] = v;
  __syncthreads();
  if (w == 0) {
    float s = (threadIdx.x < 8) ? red[threadIdx.x] : 0.f;
    for (int m = 4; m; m >>= 1) s += __shfl_xor(s, m, 32);
    if (threadIdx.x == 0) red[0] = s;
  }
  __syncthreads();
  return red[0];
}

__global__ __launch_bounds__(256) void final_kernel(
    const float* __restrict__ tk, const float* __restrict__ agg,
    float* __restrict__ out) {
  const int b = blockIdx.x, tid = threadIdx.x;
  __shared__ float red[8];
  __shared__ float colnorm[64];

  const float tv = tk[b * 256 + tid];
  const float tss = block_reduce_sum(tv * tv, red);
  const float tkn = fmaxf(sqrtf(tss), 1e-12f);

  if (tid < 64) {
    float cs = 0.f;
    for (int l = 0; l < 128; ++l) {
      const float x = agg[(size_t)b * 8192 + l * 64 + tid];
      cs += x * x;
    }
    colnorm[tid] = fmaxf(sqrtf(cs), 1e-12f);
  }
  __syncthreads();

  const float tvn = tv / tkn;
  float part = tvn * tvn;
  for (int e = tid; e < 8192; e += 256) {
    const float x = agg[(size_t)b * 8192 + e] / colnorm[e & 63];
    part += x * x;
  }
  const float tot = block_reduce_sum(part, red);
  const float rt = 1.f / fmaxf(sqrtf(tot), 1e-12f);

  out[(size_t)b * 8448 + tid] = tvn * rt;
  for (int e = tid; e < 8192; e += 256)
    out[(size_t)b * 8448 + 256 + e] =
        (agg[(size_t)b * 8192 + e] / colnorm[e & 63]) * rt;
}

// ---------------------------------------------------------------------------
extern "C" void kernel_launch(void* const* d_in, const int* in_sizes, int n_in,
                              void* d_out, int out_size, void* d_ws,
                              size_t ws_size, hipStream_t stream) {
  const float* x    = (const float*)d_in[0];
  const float* t    = (const float*)d_in[1];
  const float* Wc1  = (const float*)d_in[2];
  const float* bc1  = (const float*)d_in[3];
  const float* Wc2  = (const float*)d_in[4];
  const float* bc2  = (const float*)d_in[5];
  const float* Ws1  = (const float*)d_in[6];
  const float* bs1  = (const float*)d_in[7];
  const float* Ws2  = (const float*)d_in[8];
  const float* bs2  = (const float*)d_in[9];
  const float* Wt1  = (const float*)d_in[10];
  const float* bt1  = (const float*)d_in[11];
  const float* Wt2  = (const float*)d_in[12];
  const float* bt2  = (const float*)d_in[13];
  const float* dust = (const float*)d_in[14];
  const float* ba   = (const float*)d_in[15];
  const float* bb   = (const float*)d_in[16];
  const float* bp   = (const float*)d_in[17];
  float* out = (float*)d_out;

  char* w = (char*)d_ws;
  auto carve = [&](size_t bytes) {
    void* p = (void*)w;
    w += (bytes + 255) & ~(size_t)255;
    return p;
  };
  _Float16* hF16 = (_Float16*)carve((size_t)32 * 512 * 1024 * 2);  // h, then hs
  float*    fF32 = (float*)carve((size_t)32 * 128 * 1024 * 4);
  float*    pF32 = (float*)carve((size_t)32 * 64 * 1024 * 4);
  _Float16* fn   = (_Float16*)carve((size_t)32 * 128 * 1024 * 2);
  _Float16* fnT  = (_Float16*)carve((size_t)32 * 1024 * 128 * 2);
  float*    bwv  = (float*)carve((size_t)32 * 1024 * 4);
  float*    u    = (float*)carve((size_t)32 * 65 * 4);
  float*    v    = (float*)carve((size_t)32 * 1024 * 4);
  _Float16* Pt   = (_Float16*)carve((size_t)32 * 1024 * 64 * 2);
  float*    aggF = (float*)carve((size_t)32 * 128 * 64 * 4);
  float*    tk   = (float*)carve((size_t)32 * 256 * 4);

  const dim3 blk(256);

  // 1) h = relu(Wc1 @ x + bc1)   [M=512,N=1024,K=1536], out f16
  gemm_wmma<64, 128, true, float, float><<<dim3(8, 8, 32), blk, 0, stream>>>(
      Wc1, 0ull, x, 1572864ull, bc1, nullptr, hF16, 512, 1024, 1536);
  // 2) f = Wc2 @ h + bc2         [128,1024,512], out f32
  gemm_wmma<64, 128, false, float, _Float16><<<dim3(8, 2, 32), blk, 0, stream>>>(
      Wc2, 0ull, hF16, 524288ull, bc2, fF32, nullptr, 128, 1024, 512);
  // 3) hs = relu(Ws1 @ x + bs1)  (reuse h buffer)
  gemm_wmma<64, 128, true, float, float><<<dim3(8, 8, 32), blk, 0, stream>>>(
      Ws1, 0ull, x, 1572864ull, bs1, nullptr, hF16, 512, 1024, 1536);
  // 4) p = Ws2 @ hs + bs2        [64,1024,512], out f32
  gemm_wmma<64, 128, false, float, _Float16><<<dim3(8, 1, 32), blk, 0, stream>>>(
      Ws2, 0ull, hF16, 524288ull, bs2, pF32, nullptr, 64, 1024, 512);
  // 5) token MLP
  token_mlp<<<dim3(32), blk, 0, stream>>>(t, Wt1, bt1, Wt2, bt2, tk);
  // 6) Sinkhorn
  sinkhorn_kernel<<<dim3(32), blk, 0, stream>>>(pF32, dust, u, v);
  // 7) normalize f -> fn, fnT (f16)
  normalize_f<<<dim3(4, 32), blk, 0, stream>>>(fF32, fn, fnT);
  // 8) burst row sums (fused WMMA + sigmoid)
  burst_kernel<<<dim3(8, 32), blk, 0, stream>>>(fn, fnT, ba, bb, bwv);
  // 9) P (transposed, f16, burst-divided)
  finalize_P<<<dim3(4, 32), blk, 0, stream>>>(pF32, u, v, bwv, bp, Pt);
  // 10) agg = f @ Pt             [M=128,N=64,K=1024] per batch
  gemm_wmma<128, 64, false, float, _Float16><<<dim3(1, 1, 32), blk, 0, stream>>>(
      fF32, 131072ull, Pt, 65536ull, (const float*)nullptr, aggF, nullptr,
      128, 64, 1024);
  // 11) normalize + concat + global L2
  final_kernel<<<dim3(32), blk, 0, stream>>>(tk, aggF, out);
}
